// SlotODEFunc_10256381903328
// MI455X (gfx1250) — matrix operand order repaired and checked by reading, not compiled, not run on hardware
//
#include <hip/hip_runtime.h>
#include <hip/hip_bf16.h>
#include <math.h>

typedef __attribute__((ext_vector_type(16))) __bf16 v16bf;
typedef __attribute__((ext_vector_type(8)))  float  v8f;

#define DEV __device__ __forceinline__

// ---------------- problem constants ----------------
#define B_SZ   32
#define NSLOT  128
#define NFEAT  4096
#define SDIM   256
#define HID    1024
#define DEMB   64
#define NFREQ  128
#define SINDIM 257
#define MCHUNKS 8
#define MPERCHUNK (NFEAT / MCHUNKS)   // 512
#define MT     32                     // columns per softmax panel
#define NIT    (MPERCHUNK / MT)       // 16 inner iterations
#define PANEL_LD 36                   // padded f32 row (conflict-free)

// ---------------- dynamic LDS layout for k_attn (bytes) ----------------
#define KSTAGE_OFF  0                          // 32x256 f32 = 32768
#define VSTAGE_OFF  32768                      // 32x256 f32 = 32768
#define KBF_OFF     65536                      // 32 x KBF_LD bf16
#define KBF_LD      264                        // 528B row stride (16B mult)
#define VT_OFF      (KBF_OFF + 32 * KBF_LD * 2)        // 82432
#define VT_LD       40                         // 80B row stride (16B mult)
#define PANEL_OFF   (VT_OFF + 256 * VT_LD * 2)         // 102912
#define COLSUM_OFF  (PANEL_OFF + 128 * PANEL_LD * 4)   // 121344
#define ROWSUML_OFF (COLSUM_OFF + 128)                 // 121472
#define SMEM_BYTES  (ROWSUML_OFF + 512)                // 121984

// ---------------- workspace layout (bytes) ----------------
constexpr size_t SZ_HVEC   = 4 * DEMB * sizeof(float);
constexpr size_t SZ_WQ     = (size_t)SDIM * SDIM * 2;       // bf16
constexpr size_t SZ_WG     = (size_t)SDIM * 2 * SDIM * 2;
constexpr size_t SZ_WF0    = (size_t)HID * 2 * SDIM * 2;
constexpr size_t SZ_WF1    = (size_t)SDIM * HID * 2;
constexpr size_t SZ_BF     = (size_t)B_SZ * NSLOT * SDIM * 2;  // bf16 [32,128,256]
constexpr size_t SZ_H      = (size_t)B_SZ * NSLOT * HID * 2;   // bf16 [32,128,1024]
constexpr size_t SZ_FACC   = (size_t)B_SZ * NSLOT * SDIM * 4;  // f32
constexpr size_t SZ_ROWSUM = (size_t)B_SZ * NSLOT * 4;

constexpr size_t OFF_HVEC  = 0;
constexpr size_t OFF_WQ    = OFF_HVEC  + SZ_HVEC;
constexpr size_t OFF_WG    = OFF_WQ    + SZ_WQ;
constexpr size_t OFF_WF0   = OFF_WG    + SZ_WG;
constexpr size_t OFF_WF1   = OFF_WF0   + SZ_WF0;
constexpr size_t OFF_NORMA = OFF_WF1   + SZ_WF1;
constexpr size_t OFF_NORMF = OFF_NORMA + SZ_BF;
constexpr size_t OFF_Q     = OFF_NORMF + SZ_BF;
constexpr size_t OFF_FATTN = OFF_Q     + SZ_BF;
constexpr size_t OFF_H     = OFF_FATTN + SZ_BF;
constexpr size_t OFF_FACC  = OFF_H     + SZ_H;
constexpr size_t OFF_ROWSUM= OFF_FACC  + SZ_FACC;

// ---------------- CDNA5 async copy (global -> LDS, ASYNCcnt-tracked) --------
DEV void async_g2l_b128(unsigned lds_off, const float* gsrc) {
  asm volatile("global_load_async_to_lds_b128 %0, %1, off"
               :: "v"(lds_off), "v"(gsrc) : "memory");
}
DEV void wait_async0() {
  asm volatile("s_wait_asynccnt 0x0" ::: "memory");
}

// ---------------- WMMA helpers (ISA 7.12.2 layouts) ----------------
DEV v8f wmma_bf16(v16bf a, v16bf b, v8f c) {
  return __builtin_amdgcn_wmma_f32_16x16x32_bf16(false, a, false, b, (short)0, c,
                                                 false, false);
}

// A fragment 16x32 bf16, row-major src, row stride lda elements.
DEV v16bf load_a_bf16(const __bf16* A, int lda, int row0, int k0, int lane) {
  const int m = lane & 15;
  const int half = lane >> 4;
  const __bf16* base = A + (size_t)(row0 + m) * lda + k0;
  v16bf a;
#pragma unroll
  for (int j = 0; j < 4; ++j) {
    const int kk = half * 8 + 2 * j;
    a[2 * j]     = base[kk];
    a[2 * j + 1] = base[kk + 1];
    a[8 + 2 * j]     = base[16 + kk];
    a[8 + 2 * j + 1] = base[16 + kk + 1];
  }
  return a;
}

// Same pattern, converting from f32 (LDS att panel).
DEV v16bf load_a_f32cvt(const float* P, int ldp, int row0, int k0, int lane) {
  const int m = lane & 15;
  const int half = lane >> 4;
  const float* base = P + (size_t)(row0 + m) * ldp + k0;
  v16bf a;
#pragma unroll
  for (int j = 0; j < 4; ++j) {
    const int kk = half * 8 + 2 * j;
    a[2 * j]     = (__bf16)base[kk];
    a[2 * j + 1] = (__bf16)base[kk + 1];
    a[8 + 2 * j]     = (__bf16)base[16 + kk];
    a[8 + 2 * j + 1] = (__bf16)base[16 + kk + 1];
  }
  return a;
}

// B fragment 32x16 bf16 where B[kk][n] = S[(n0+n)*lda + k0+kk] (transposed read).
DEV v16bf load_bT_bf16(const __bf16* S, int lda, int n0, int k0, int lane) {
  const int n = lane & 15;
  const int ks = (lane >> 4) * 8;
  const __bf16* base = S + (size_t)(n0 + n) * lda + k0 + ks;
  v16bf b;
#pragma unroll
  for (int i = 0; i < 16; ++i) b[i] = base[i < 8 ? i : i + 8];
  return b;
}

DEV void store_d_bf16(__bf16* D, int ldd, int row0, int col0, int lane, v8f acc) {
  const int n = lane & 15;
  const int mofs = (lane >> 4) * 8;
#pragma unroll
  for (int r = 0; r < 8; ++r)
    D[(size_t)(row0 + r + mofs) * ldd + col0 + n] = (__bf16)acc[r];
}

DEV float silu_f(float x) { return x / (1.0f + __expf(-x)); }

// ============ kernel 1: time-dep-weight hidden vectors (tiny) ============
__global__ void __launch_bounds__(256) k_tdw_hidden(
    const float* t,
    const float* qw0, const float* qb0, const float* qw1, const float* qb1,
    const float* gw0, const float* gb0, const float* gw1, const float* gb1,
    const float* f0w0, const float* f0b0, const float* f0w1, const float* f0b1,
    const float* f1w0, const float* f1b0, const float* f1w1, const float* f1b1,
    float* hvec)
{
  __shared__ float h0s[4][DEMB];
  const int tid = threadIdx.x;
  const int p = tid >> 6, j = tid & 63;
  const float tv = t[0];
  const float* w0 = (p == 0) ? qw0 : (p == 1) ? gw0 : (p == 2) ? f0w0 : f1w0;
  const float* b0 = (p == 0) ? qb0 : (p == 1) ? gb0 : (p == 2) ? f0b0 : f1b0;
  const float* w1 = (p == 0) ? qw1 : (p == 1) ? gw1 : (p == 2) ? f0w1 : f1w1;
  const float* b1 = (p == 0) ? qb1 : (p == 1) ? gb1 : (p == 2) ? f0b1 : f1b1;

  const float* r0 = w0 + (size_t)j * SINDIM;
  float acc = b0[j] + r0[0] * tv;
  const float fstep = -0.07195578415625f;  // -log(10000)/128
  for (int i = 0; i < NFREQ; ++i) {
    const float wt = tv * (fstep * (float)i);
    acc += r0[1 + i] * __sinf(wt) + r0[1 + NFREQ + i] * __cosf(wt);
  }
  h0s[p][j] = silu_f(acc);
  __syncthreads();

  const float* r1 = w1 + (size_t)j * DEMB;
  float a1 = b1[j];
  for (int k = 0; k < DEMB; ++k) a1 += r1[k] * h0s[p][k];
  hvec[p * DEMB + j] = silu_f(a1);
}

// ============ kernel 2: materialize W = wp @ h + bp (memory bound) ============
__global__ void __launch_bounds__(256) k_tdw_weights(
    const float* qwp, const float* qbp, const float* gwp, const float* gbp,
    const float* f0wp, const float* f0bp, const float* f1wp, const float* f1bp,
    const float* hvec,
    __bf16* Wq, __bf16* Wg, __bf16* Wf0, __bf16* Wf1)
{
  __shared__ float hs[4][DEMB];
  hs[threadIdx.x >> 6][threadIdx.x & 63] = hvec[threadIdx.x & 255];
  __syncthreads();
  const int total = 65536 + 131072 + 524288 + 262144;  // 983040
  for (int o = blockIdx.x * blockDim.x + threadIdx.x; o < total;
       o += gridDim.x * blockDim.x) {
    const float *wp, *bp; __bf16* dst; int lo, head;
    if (o < 65536)        { head = 0; lo = o;          wp = qwp;  bp = qbp;  dst = Wq;  }
    else if (o < 196608)  { head = 1; lo = o - 65536;  wp = gwp;  bp = gbp;  dst = Wg;  }
    else if (o < 720896)  { head = 2; lo = o - 196608; wp = f0wp; bp = f0bp; dst = Wf0; }
    else                  { head = 3; lo = o - 720896; wp = f1wp; bp = f1bp; dst = Wf1; }
    const float4* row = (const float4*)(wp + (size_t)lo * DEMB);
    float acc = bp[lo];
#pragma unroll
    for (int i = 0; i < 16; ++i) {
      const float4 w = row[i];
      acc += w.x * hs[head][4 * i]     + w.y * hs[head][4 * i + 1] +
             w.z * hs[head][4 * i + 2] + w.w * hs[head][4 * i + 3];
    }
    dst[lo] = (__bf16)acc;
  }
}

// ============ kernel 3: LayerNorms + q = normA @ Wq^T (WMMA) ============
__global__ void __launch_bounds__(256) k_norm_q(
    const float* slots, const float* lnAw, const float* lnAb,
    const float* lnFw, const float* lnFb, const __bf16* Wq,
    __bf16* normA, __bf16* normF, __bf16* qout)
{
  const int b = blockIdx.x;
  const int tid = threadIdx.x;
  if (tid < NSLOT) {
    const float* srow = slots + ((size_t)b * NSLOT + tid) * SDIM;
    float s = 0.f;
    for (int c = 0; c < SDIM; ++c) s += srow[c];
    const float mu = s * (1.0f / SDIM);
    float v = 0.f;
    for (int c = 0; c < SDIM; ++c) { const float d = srow[c] - mu; v += d * d; }
    const float rinv = rsqrtf(v * (1.0f / SDIM) + 1e-5f);
    __bf16* na = normA + ((size_t)b * NSLOT + tid) * SDIM;
    __bf16* nf = normF + ((size_t)b * NSLOT + tid) * SDIM;
    for (int c = 0; c < SDIM; ++c) {
      const float xn = (srow[c] - mu) * rinv;
      na[c] = (__bf16)(xn * lnAw[c] + lnAb[c]);
      nf[c] = (__bf16)(xn * lnFw[c] + lnFb[c]);
    }
  }
  __threadfence_block();
  __syncthreads();

  const int wave = tid >> 5, lane = tid & 31;
  const int row0 = wave * 16;
  const __bf16* Ab = normA + (size_t)b * NSLOT * SDIM;
  __bf16* qb = qout + (size_t)b * NSLOT * SDIM;
  for (int dt = 0; dt < 16; ++dt) {
    v8f acc = {};
#pragma unroll
    for (int kf = 0; kf < 8; ++kf) {
      const v16bf a  = load_a_bf16(Ab, SDIM, row0, kf * 32, lane);
      const v16bf bb = load_bT_bf16(Wq, SDIM, dt * 16, kf * 32, lane);
      acc = wmma_bf16(a, bb, acc);
    }
    store_d_bf16(qb, SDIM, row0, dt * 16, lane, acc);
  }
}

// ====== kernel 4: async-staged logits (WMMA) + slot-softmax + att@v (WMMA) ======
__global__ void __launch_bounds__(256) k_attn(
    const float* kmat, const float* vmat, const __bf16* qbf,
    float* facc, float* rowsum)
{
  extern __shared__ char smem[];
  float*  kstage  = (float*)(smem + KSTAGE_OFF);   // raw f32 tile (async dest)
  float*  vstage  = (float*)(smem + VSTAGE_OFF);
  __bf16* kbf     = (__bf16*)(smem + KBF_OFF);     // [32][KBF_LD] bf16
  __bf16* vT      = (__bf16*)(smem + VT_OFF);      // [256][VT_LD] bf16 (transposed)
  float*  panel   = (float*)(smem + PANEL_OFF);    // [128][PANEL_LD] f32
  float*  colsum  = (float*)(smem + COLSUM_OFF);   // [MT]
  float*  rowsumL = (float*)(smem + ROWSUML_OFF);  // [128]

  const int b = blockIdx.x >> 3;           // MCHUNKS == 8
  const int chunk = blockIdx.x & 7;
  const int tid = threadIdx.x;
  const int wave = tid >> 5, lane = tid & 31;
  const int row0 = wave * 16;
  const int n = lane & 15, mofs = (lane >> 4) * 8;
  const float* kb = kmat + (size_t)b * NFEAT * SDIM;
  const float* vb = vmat + (size_t)b * NFEAT * SDIM;
  const __bf16* qb = qbf + (size_t)b * NSLOT * SDIM;

  const unsigned kstage_lds = (unsigned)(uintptr_t)(void*)kstage;
  const unsigned vstage_lds = (unsigned)(uintptr_t)(void*)vstage;

  // q A-fragments for this wave's 16 slot rows (K = 256), held in VGPRs
  v16bf qA[8];
#pragma unroll
  for (int kf = 0; kf < 8; ++kf) qA[kf] = load_a_bf16(qb, SDIM, row0, kf * 32, lane);

  v8f fac[16];
#pragma unroll
  for (int dt = 0; dt < 16; ++dt)
#pragma unroll
    for (int r = 0; r < 8; ++r) fac[dt][r] = 0.f;

  const int m0base = chunk * MPERCHUNK;

  // ---- prologue: async-prefetch tile 0 (k and v, each 32KB contiguous) ----
#pragma unroll
  for (int i = 0; i < 8; ++i) {
    const int c = tid + i * 256;           // 16-byte chunk index, 0..2047
    async_g2l_b128(kstage_lds + c * 16, kb + (size_t)m0base * SDIM + c * 4);
    async_g2l_b128(vstage_lds + c * 16, vb + (size_t)m0base * SDIM + c * 4);
  }
  if (tid < NSLOT) rowsumL[tid] = 0.f;

  for (int it = 0; it < NIT; ++it) {
    const int m0 = m0base + it * MT;
    // wait for this tile's async copies (per-wave), then sync all waves
    wait_async0();
    __syncthreads();

    // ---- cooperative f32 -> bf16 conversion (once per WG, not per wave) ----
#pragma unroll
    for (int i = 0; i < 8; ++i) {
      const int c = tid + i * 256;         // float4 chunk, 0..2047
      const int e = c * 4;
      {
        const float4 x = ((const float4*)kstage)[c];
        __bf16* d = kbf + (e >> 8) * KBF_LD + (e & 255);
        d[0] = (__bf16)x.x; d[1] = (__bf16)x.y;
        d[2] = (__bf16)x.z; d[3] = (__bf16)x.w;
      }
      {
        const float4 x = ((const float4*)vstage)[c];
        const int m = e >> 8, dd = e & 255;
        vT[(dd + 0) * VT_LD + m] = (__bf16)x.x;
        vT[(dd + 1) * VT_LD + m] = (__bf16)x.y;
        vT[(dd + 2) * VT_LD + m] = (__bf16)x.z;
        vT[(dd + 3) * VT_LD + m] = (__bf16)x.w;
      }
    }
    __syncthreads();

    // ---- async-prefetch next tile while this one is computed ----
    if (it + 1 < NIT) {
      const size_t mnxt = (size_t)(m0 + MT) * SDIM;
#pragma unroll
      for (int i = 0; i < 8; ++i) {
        const int c = tid + i * 256;
        async_g2l_b128(kstage_lds + c * 16, kb + mnxt + c * 4);
        async_g2l_b128(vstage_lds + c * 16, vb + mnxt + c * 4);
      }
    }

    // ---- logits panel: two 16-col WMMA tiles per wave, K = 256, B from LDS ----
#pragma unroll
    for (int half = 0; half < 2; ++half) {
      v8f lg = {};
#pragma unroll
      for (int kf = 0; kf < 8; ++kf) {
        const v16bf bb = load_bT_bf16(kbf, KBF_LD, half * 16, kf * 32, lane);
        lg = wmma_bf16(qA[kf], bb, lg);
      }
#pragma unroll
      for (int r = 0; r < 8; ++r)
        panel[(row0 + r + mofs) * PANEL_LD + half * 16 + n] = lg[r] * 0.0625f;
    }
    __syncthreads();

    // ---- softmax over the slot axis (each of 32 columns independent) ----
    if (tid < MT) {
      const int c = tid;
      float mx = -3.4e38f;
      for (int r = 0; r < NSLOT; ++r) mx = fmaxf(mx, panel[r * PANEL_LD + c]);
      float s = 0.f;
      for (int r = 0; r < NSLOT; ++r) {
        const float e = __expf(panel[r * PANEL_LD + c] - mx);
        panel[r * PANEL_LD + c] = e; s += e;
      }
      colsum[c] = s;
    }
    __syncthreads();
    if (tid < NSLOT) {
      float rs = 0.f;
      for (int c = 0; c < MT; ++c) {
        const float a = panel[tid * PANEL_LD + c] / colsum[c];
        panel[tid * PANEL_LD + c] = a; rs += a;
      }
      rowsumL[tid] += rs;
    }
    __syncthreads();

    // ---- facc += att_panel(16x32) @ v(32x256), B from transposed LDS tile ----
    const v16bf attA = load_a_f32cvt(panel, PANEL_LD, row0, 0, lane);
#pragma unroll
    for (int dt = 0; dt < 16; ++dt) {
      const v16bf bb = load_bT_bf16(vT, VT_LD, dt * 16, 0, lane);
      fac[dt] = wmma_bf16(attA, bb, fac[dt]);
    }
    __syncthreads();
  }

  float* fb = facc + (size_t)b * NSLOT * SDIM;
#pragma unroll
  for (int dt = 0; dt < 16; ++dt)
#pragma unroll
    for (int r = 0; r < 8; ++r)
      unsafeAtomicAdd(&fb[(size_t)(row0 + r + mofs) * SDIM + dt * 16 + n], fac[dt][r]);
  if (tid < NSLOT) unsafeAtomicAdd(&rowsum[b * NSLOT + tid], rowsumL[tid]);
}

// ====== kernel 5: renorm, gate GEMM, FF0/FF1 GEMMs, combine (WMMA) ======
__global__ void __launch_bounds__(256) k_final(
    const float* facc, const float* rowsum,
    const __bf16* normA, const __bf16* normF,
    const __bf16* Wg, const __bf16* Wf0, const __bf16* Wf1,
    __bf16* fattn, __bf16* hbuf, float* out)
{
  const int b = blockIdx.x;
  const int tid = threadIdx.x;
  const int wave = tid >> 5, lane = tid & 31;
  const int row0 = wave * 16;
  const int n = lane & 15, mofs = (lane >> 4) * 8;
  const float* fb  = facc + (size_t)b * NSLOT * SDIM;
  const float* rsb = rowsum + (size_t)b * NSLOT;
  const __bf16* nAb = normA + (size_t)b * NSLOT * SDIM;
  const __bf16* nFb = normF + (size_t)b * NSLOT * SDIM;
  __bf16* fab = fattn + (size_t)b * NSLOT * SDIM;
  __bf16* hb  = hbuf + (size_t)b * NSLOT * HID;
  float* ob   = out + (size_t)b * NSLOT * SDIM;

  // f_attn = facc/(rowsum+eps) -> bf16, for this wave's 16 rows only
  for (int idx = lane; idx < 16 * SDIM; idx += 32) {
    const int r = idx >> 8, c = idx & 255;
    const size_t g = (size_t)(row0 + r) * SDIM + c;
    fab[g] = (__bf16)(fb[g] / (rsb[row0 + r] + 1e-8f));
  }

  // gate = sigmoid([normA|f_attn] @ Wg^T); out = gate * f_attn (pass 1)
  for (int dt = 0; dt < 16; ++dt) {
    v8f acc = {};
    for (int kf = 0; kf < 16; ++kf) {
      const __bf16* Asrc = (kf < 8) ? nAb : fab;
      const int k0 = (kf < 8) ? kf * 32 : (kf - 8) * 32;
      const v16bf a  = load_a_bf16(Asrc, SDIM, row0, k0, lane);
      const v16bf bb = load_bT_bf16(Wg, 2 * SDIM, dt * 16, kf * 32, lane);
      acc = wmma_bf16(a, bb, acc);
    }
#pragma unroll
    for (int r = 0; r < 8; ++r) {
      const int rr = row0 + r + mofs, cc = dt * 16 + n;
      const float fa = fb[(size_t)rr * SDIM + cc] / (rsb[rr] + 1e-8f);
      const float g = 1.0f / (1.0f + __expf(-acc[r]));
      ob[(size_t)rr * SDIM + cc] = g * fa;
    }
  }

  // h = relu([normF|f_attn] @ Wf0^T) -> bf16 hbuf
  for (int ht = 0; ht < 64; ++ht) {
    v8f acc = {};
    for (int kf = 0; kf < 16; ++kf) {
      const __bf16* Asrc = (kf < 8) ? nFb : fab;
      const int k0 = (kf < 8) ? kf * 32 : (kf - 8) * 32;
      const v16bf a  = load_a_bf16(Asrc, SDIM, row0, k0, lane);
      const v16bf bb = load_bT_bf16(Wf0, 2 * SDIM, ht * 16, kf * 32, lane);
      acc = wmma_bf16(a, bb, acc);
    }
#pragma unroll
    for (int r = 0; r < 8; ++r)
      hb[(size_t)(row0 + r + mofs) * HID + ht * 16 + n] = (__bf16)fmaxf(acc[r], 0.f);
  }

  // out += h @ Wf1^T (pass 2)
  for (int dt = 0; dt < 16; ++dt) {
    v8f acc = {};
    for (int kf = 0; kf < 32; ++kf) {
      const v16bf a  = load_a_bf16(hb, HID, row0, kf * 32, lane);
      const v16bf bb = load_bT_bf16(Wf1, HID, dt * 16, kf * 32, lane);
      acc = wmma_bf16(a, bb, acc);
    }
#pragma unroll
    for (int r = 0; r < 8; ++r) {
      const int rr = row0 + r + mofs, cc = dt * 16 + n;
      ob[(size_t)rr * SDIM + cc] += acc[r];
    }
  }
}

// ---------------- launch ----------------
extern "C" void kernel_launch(void* const* d_in, const int* in_sizes, int n_in,
                              void* d_out, int out_size, void* d_ws, size_t ws_size,
                              hipStream_t stream) {
  (void)in_sizes; (void)n_in; (void)out_size; (void)ws_size;
  const float* qw0  = (const float*)d_in[0];
  const float* qb0  = (const float*)d_in[1];
  const float* qw1  = (const float*)d_in[2];
  const float* qb1  = (const float*)d_in[3];
  const float* qwp  = (const float*)d_in[4];
  const float* qbp  = (const float*)d_in[5];
  const float* gw0  = (const float*)d_in[6];
  const float* gb0  = (const float*)d_in[7];
  const float* gw1  = (const float*)d_in[8];
  const float* gb1  = (const float*)d_in[9];
  const float* gwp  = (const float*)d_in[10];
  const float* gbp  = (const float*)d_in[11];
  const float* f0w0 = (const float*)d_in[12];
  const float* f0b0 = (const float*)d_in[13];
  const float* f0w1 = (const float*)d_in[14];
  const float* f0b1 = (const float*)d_in[15];
  const float* f0wp = (const float*)d_in[16];
  const float* f0bp = (const float*)d_in[17];
  const float* f1w0 = (const float*)d_in[18];
  const float* f1b0 = (const float*)d_in[19];
  const float* f1w1 = (const float*)d_in[20];
  const float* f1b1 = (const float*)d_in[21];
  const float* f1wp = (const float*)d_in[22];
  const float* f1bp = (const float*)d_in[23];
  const float* lnAw = (const float*)d_in[24];
  const float* lnAb = (const float*)d_in[25];
  const float* lnFw = (const float*)d_in[26];
  const float* lnFb = (const float*)d_in[27];
  const float* t    = (const float*)d_in[28];
  const float* slots= (const float*)d_in[29];
  const float* kmat = (const float*)d_in[30];
  const float* vmat = (const float*)d_in[31];

  char* ws = (char*)d_ws;
  float* hvec    = (float*)(ws + OFF_HVEC);
  __bf16* Wq     = (__bf16*)(ws + OFF_WQ);
  __bf16* Wg     = (__bf16*)(ws + OFF_WG);
  __bf16* Wf0    = (__bf16*)(ws + OFF_WF0);
  __bf16* Wf1    = (__bf16*)(ws + OFF_WF1);
  __bf16* normA  = (__bf16*)(ws + OFF_NORMA);
  __bf16* normF  = (__bf16*)(ws + OFF_NORMF);
  __bf16* qbuf   = (__bf16*)(ws + OFF_Q);
  __bf16* fattn  = (__bf16*)(ws + OFF_FATTN);
  __bf16* hbuf   = (__bf16*)(ws + OFF_H);
  float* facc    = (float*)(ws + OFF_FACC);
  float* rowsum  = (float*)(ws + OFF_ROWSUM);
  float* out     = (float*)d_out;

  hipMemsetAsync(ws + OFF_FACC, 0, SZ_FACC + SZ_ROWSUM, stream);

  k_tdw_hidden<<<1, 256, 0, stream>>>(t,
      qw0, qb0, qw1, qb1, gw0, gb0, gw1, gb1,
      f0w0, f0b0, f0w1, f0b1, f1w0, f1b0, f1w1, f1b1, hvec);

  k_tdw_weights<<<3840, 256, 0, stream>>>(
      qwp, qbp, gwp, gbp, f0wp, f0bp, f1wp, f1bp, hvec, Wq, Wg, Wf0, Wf1);

  k_norm_q<<<B_SZ, 256, 0, stream>>>(
      slots, lnAw, lnAb, lnFw, lnFb, Wq, normA, normF, qbuf);

  k_attn<<<B_SZ * MCHUNKS, 256, SMEM_BYTES, stream>>>(kmat, vmat, qbuf, facc, rowsum);

  k_final<<<B_SZ, 256, 0, stream>>>(
      facc, rowsum, normA, normF, Wg, Wf0, Wf1, fattn, hbuf, out);
}
